// MultiHeadAttention_78039555768334
// MI455X (gfx1250) — matrix operand order repaired and checked by reading, not compiled, not run on hardware
//
#include <hip/hip_runtime.h>
#include <hip/hip_bf16.h>

typedef __attribute__((ext_vector_type(8)))  __bf16 v8bf;
typedef __attribute__((ext_vector_type(16))) __bf16 v16bf;
typedef __attribute__((ext_vector_type(8)))  float  v8f;
typedef __attribute__((ext_vector_type(4)))  int    v4i;

#define DMODEL 1024
#define NHEAD  16
#define DK     64
#define SEQ    2048
#define BATCH  4
#define MROWS  (BATCH*SEQ)   // 8192

#if defined(__AMDGCN__) && __has_builtin(__builtin_amdgcn_global_load_async_to_lds_b128)
#define HAVE_ASYNC_LDS 1
#else
#define HAVE_ASYNC_LDS 0
#endif

__device__ __forceinline__ void cp_async16(void* lds, const void* g) {
#if HAVE_ASYNC_LDS
  __builtin_amdgcn_global_load_async_to_lds_b128(
      (__attribute__((address_space(1))) v4i*)g,
      (__attribute__((address_space(3))) v4i*)lds, 0, 0);
#else
  *(uint4*)lds = *(const uint4*)g;   // synchronous fallback
#endif
}

__device__ __forceinline__ void wait_async_le(bool steady) {
#if HAVE_ASYNC_LDS
#if __has_builtin(__builtin_amdgcn_s_wait_asynccnt)
  if (steady) __builtin_amdgcn_s_wait_asynccnt(4);
  else        __builtin_amdgcn_s_wait_asynccnt(0);
#else
  if (steady) asm volatile("s_wait_asynccnt 4" ::: "memory");
  else        asm volatile("s_wait_asynccnt 0" ::: "memory");
#endif
#else
  (void)steady;
#endif
}

__device__ __forceinline__ v16bf cat16(v8bf lo, v8bf hi) {
  v16bf r;
#pragma unroll
  for (int i = 0; i < 8; ++i) { r[i] = lo[i]; r[i + 8] = hi[i]; }
  return r;
}

__device__ __forceinline__ v8f wmma_bf16(v16bf a, v16bf b, v8f c) {
  return __builtin_amdgcn_wmma_f32_16x16x32_bf16(
      false, a, false, b, (short)0, c, false, false);
}

// ---------------------------------------------------------------------------
// fp32 -> bf16 conversion (weights)
// ---------------------------------------------------------------------------
__global__ void cvt_f32_to_bf16(const float* __restrict__ in,
                                __bf16* __restrict__ out, int n) {
  int i = blockIdx.x * blockDim.x + threadIdx.x;
  if (i < n) out[i] = (__bf16)in[i];
}

// ---------------------------------------------------------------------------
// GEMM: C[M,N] = (A[M,K] * W[N,K]^T + bias[N]) * oscale
//   OUT_MODE: 0 = bf16 [M,N]; 1 = bf16 -> [B,H,DK,S] (V transposed); 2 = fp32
// Block: 256 thr = 8 waves, tile 128x128; wave tile 32(M) x 64(N).
// ---------------------------------------------------------------------------
template <bool A_F32, int OUT_MODE>
__global__ __launch_bounds__(256) void gemm_wmma(
    const void* __restrict__ Av, const __bf16* __restrict__ W,
    const float* __restrict__ bias, void* __restrict__ Out,
    int M, int N, int K, float oscale)
{
  const int lane = threadIdx.x & 31;
  const int wave = threadIdx.x >> 5;
  const int l16  = lane & 15;
  const int hi   = lane >> 4;
  const int mbase = blockIdx.y * 128 + (wave & 3) * 32;
  const int nbase = blockIdx.x * 128 + (wave >> 2) * 64;

  v8f acc[2][4] = {};

  for (int k0 = 0; k0 < K; k0 += 32) {
    v16bf af[2];
#pragma unroll
    for (int mt = 0; mt < 2; ++mt) {
      int row = mbase + mt * 16 + l16;
      if (A_F32) {
        const float* ap = (const float*)Av + (size_t)row * K + k0 + hi * 8;
        float4 x0 = *(const float4*)(ap);
        float4 x1 = *(const float4*)(ap + 4);
        float4 y0 = *(const float4*)(ap + 16);
        float4 y1 = *(const float4*)(ap + 20);
        v16bf a;
        a[0]  = (__bf16)x0.x; a[1]  = (__bf16)x0.y; a[2]  = (__bf16)x0.z; a[3]  = (__bf16)x0.w;
        a[4]  = (__bf16)x1.x; a[5]  = (__bf16)x1.y; a[6]  = (__bf16)x1.z; a[7]  = (__bf16)x1.w;
        a[8]  = (__bf16)y0.x; a[9]  = (__bf16)y0.y; a[10] = (__bf16)y0.z; a[11] = (__bf16)y0.w;
        a[12] = (__bf16)y1.x; a[13] = (__bf16)y1.y; a[14] = (__bf16)y1.z; a[15] = (__bf16)y1.w;
        af[mt] = a;
      } else {
        const __bf16* ap = (const __bf16*)Av + (size_t)row * K + k0 + hi * 8;
        af[mt] = cat16(*(const v8bf*)ap, *(const v8bf*)(ap + 16));
      }
    }
    v16bf bfr[4];
#pragma unroll
    for (int nt = 0; nt < 4; ++nt) {
      const __bf16* wp = W + (size_t)(nbase + nt * 16 + l16) * K + k0 + hi * 8;
      bfr[nt] = cat16(*(const v8bf*)wp, *(const v8bf*)(wp + 16));
    }
#pragma unroll
    for (int mt = 0; mt < 2; ++mt)
#pragma unroll
      for (int nt = 0; nt < 4; ++nt)
        acc[mt][nt] = wmma_bf16(af[mt], bfr[nt], acc[mt][nt]);
  }

  float bv[4];
#pragma unroll
  for (int nt = 0; nt < 4; ++nt) bv[nt] = bias[nbase + nt * 16 + l16];

#pragma unroll
  for (int mt = 0; mt < 2; ++mt) {
#pragma unroll
    for (int nt = 0; nt < 4; ++nt) {
#pragma unroll
      for (int r = 0; r < 8; ++r) {
        float v = (acc[mt][nt][r] + bv[nt]) * oscale;
        int m = mbase + mt * 16 + r + 8 * hi;
        int n = nbase + nt * 16 + l16;
        if (OUT_MODE == 0) {
          ((__bf16*)Out)[(size_t)m * N + n] = (__bf16)v;
        } else if (OUT_MODE == 1) {
          int b = m >> 11, s = m & (SEQ - 1);
          int h = n >> 6,  d = n & (DK - 1);
          ((__bf16*)Out)[(((size_t)(b * NHEAD + h) * DK + d) << 11) + s] = (__bf16)v;
        } else {
          ((float*)Out)[(size_t)m * N + n] = v;
        }
      }
    }
  }
}

// ---------------------------------------------------------------------------
// Flash attention. One (b,h) per 16 blocks; each wave owns 16 query rows.
// K/V tiles (64 keys x 64) staged to LDS, double-buffered via async-to-LDS.
// Qp is pre-scaled by 1/sqrt(DK)*log2(e) so scores are in exp2 domain.
// ---------------------------------------------------------------------------
__global__ __launch_bounds__(256) void attn_wmma(
    const __bf16* __restrict__ Qp, const __bf16* __restrict__ Kp,
    const __bf16* __restrict__ Vt, __bf16* __restrict__ AO)
{
  __shared__ __align__(16) __bf16 ktile[2][64 * 64];   // [key row][dk]
  __shared__ __align__(16) __bf16 vtile[2][64 * 64];   // [dv][key col]
  __shared__ __align__(16) __bf16 plds[8][16][64];     // per-wave P tile

  const int tid  = threadIdx.x;
  const int lane = tid & 31;
  const int wave = tid >> 5;
  const int l16  = lane & 15;
  const int hi   = lane >> 4;
  const int qblk = blockIdx.x & 15;
  const int bh   = blockIdx.x >> 4;
  const int b = bh >> 4, h = bh & 15;
  const int qbase = qblk * 128 + wave * 16;

  const __bf16* kb0 = Kp + (size_t)b * SEQ * DMODEL + h * DK;
  const __bf16* vb0 = Vt + (size_t)bh * DK * SEQ;

  const int srow = tid >> 2;          // staging row 0..63
  const int scol = (tid & 3) * 16;    // staging col 0,16,32,48 (x 2B = 32B)

  auto stage = [&](int kc, int buf) {
    const __bf16* gk = kb0 + (size_t)(kc + srow) * DMODEL + scol;
    cp_async16(&ktile[buf][srow * 64 + scol],     gk);
    cp_async16(&ktile[buf][srow * 64 + scol + 8], gk + 8);
    const __bf16* gv = vb0 + (size_t)srow * SEQ + kc + scol;
    cp_async16(&vtile[buf][srow * 64 + scol],     gv);
    cp_async16(&vtile[buf][srow * 64 + scol + 8], gv + 8);
  };

  // Q fragments (A-layout), loaded once
  v16bf qf[2];
  {
    const __bf16* qp =
        Qp + ((size_t)b * SEQ + qbase + l16) * DMODEL + h * DK + hi * 8;
#pragma unroll
    for (int i = 0; i < 2; ++i)
      qf[i] = cat16(*(const v8bf*)(qp + i * 32),
                    *(const v8bf*)(qp + i * 32 + 16));
  }

  v8f acc[4] = {};
  float mrow[8], lrow[8];
#pragma unroll
  for (int r = 0; r < 8; ++r) { mrow[r] = -1e30f; lrow[r] = 0.0f; }

  stage(0, 0);

  for (int kc = 0, it = 0; kc < SEQ; kc += 64, ++it) {
    const int buf = it & 1;
    const bool hasnext = (kc + 64) < SEQ;
    if (hasnext) stage(kc + 64, buf ^ 1);
    wait_async_le(hasnext);           // previous tile's copies complete
    __syncthreads();                  // ... in every wave

    // scores: four 16x16 tiles over 64 key columns
    v8f st[4] = {};
#pragma unroll
    for (int j = 0; j < 4; ++j)
#pragma unroll
      for (int i = 0; i < 2; ++i) {
        const __bf16* kp = &ktile[buf][(j * 16 + l16) * 64 + i * 32 + hi * 8];
        st[j] = wmma_bf16(qf[i],
                          cat16(*(const v8bf*)kp, *(const v8bf*)(kp + 16)),
                          st[j]);
      }

    // online softmax (exp2 domain, scale pre-folded into Q)
    float f[8];
#pragma unroll
    for (int r = 0; r < 8; ++r) {
      float a0 = st[0][r], a1 = st[1][r], a2 = st[2][r], a3 = st[3][r];
      float mx = fmaxf(fmaxf(a0, a1), fmaxf(a2, a3));
      mx = fmaxf(mx, __shfl_xor(mx, 1, 32));
      mx = fmaxf(mx, __shfl_xor(mx, 2, 32));
      mx = fmaxf(mx, __shfl_xor(mx, 4, 32));
      mx = fmaxf(mx, __shfl_xor(mx, 8, 32));
      float mnew = fmaxf(mrow[r], mx);
      float fr = exp2f(mrow[r] - mnew);
      float p0 = exp2f(a0 - mnew);
      float p1 = exp2f(a1 - mnew);
      float p2 = exp2f(a2 - mnew);
      float p3 = exp2f(a3 - mnew);
      mrow[r] = mnew;
      lrow[r] = lrow[r] * fr + ((p0 + p1) + (p2 + p3));
      f[r] = fr;
      const int rr = r + 8 * hi;
      plds[wave][rr][l16]      = (__bf16)p0;
      plds[wave][rr][16 + l16] = (__bf16)p1;
      plds[wave][rr][32 + l16] = (__bf16)p2;
      plds[wave][rr][48 + l16] = (__bf16)p3;
    }
#pragma unroll
    for (int nt = 0; nt < 4; ++nt)
#pragma unroll
      for (int r = 0; r < 8; ++r) acc[nt][r] *= f[r];

    // reload P as A-fragments (per-wave LDS region; DS in-order per wave)
    v16bf pf[2];
#pragma unroll
    for (int c = 0; c < 2; ++c) {
      const __bf16* pp = &plds[wave][l16][c * 32 + hi * 8];
      pf[c] = cat16(*(const v8bf*)pp, *(const v8bf*)(pp + 16));
    }
    // P @ V from LDS (V already [dv][key])
#pragma unroll
    for (int nt = 0; nt < 4; ++nt)
#pragma unroll
      for (int c = 0; c < 2; ++c) {
        const __bf16* vp = &vtile[buf][(nt * 16 + l16) * 64 + c * 32 + hi * 8];
        acc[nt] = wmma_bf16(pf[c],
                            cat16(*(const v8bf*)vp, *(const v8bf*)(vp + 16)),
                            acc[nt]);
      }
    __syncthreads();                  // all reads of buf done before restage
  }

  float linv[8];
#pragma unroll
  for (int r = 0; r < 8; ++r) {
    float s = lrow[r];
    s += __shfl_xor(s, 1, 32);
    s += __shfl_xor(s, 2, 32);
    s += __shfl_xor(s, 4, 32);
    s += __shfl_xor(s, 8, 32);
    linv[r] = 1.0f / s;
  }
  __bf16* aop =
      AO + ((size_t)b * SEQ + qbase + 8 * hi) * DMODEL + h * DK + l16;
#pragma unroll
  for (int nt = 0; nt < 4; ++nt)
#pragma unroll
    for (int r = 0; r < 8; ++r)
      aop[(size_t)r * DMODEL + nt * 16] = (__bf16)(acc[nt][r] * linv[r]);
}

// ---------------------------------------------------------------------------
// Host-side orchestration
// ---------------------------------------------------------------------------
extern "C" void kernel_launch(void* const* d_in, const int* in_sizes, int n_in,
                              void* d_out, int out_size, void* d_ws,
                              size_t ws_size, hipStream_t stream)
{
  (void)in_sizes; (void)n_in; (void)out_size; (void)ws_size;
  const float* query = (const float*)d_in[0];
  const float* key   = (const float*)d_in[1];
  const float* value = (const float*)d_in[2];
  const float* Wq = (const float*)d_in[3];
  const float* bq = (const float*)d_in[4];
  const float* Wk = (const float*)d_in[5];
  const float* bk = (const float*)d_in[6];
  const float* Wv = (const float*)d_in[7];
  const float* bv = (const float*)d_in[8];
  const float* Wo = (const float*)d_in[9];
  const float* bo = (const float*)d_in[10];

  char* ws = (char*)d_ws;
  const size_t WBYTES = (size_t)DMODEL * DMODEL * sizeof(__bf16);  // 2 MB
  const size_t ABYTES = (size_t)MROWS * DMODEL * sizeof(__bf16);   // 16 MB
  __bf16* wqb = (__bf16*)(ws + 0 * WBYTES);
  __bf16* wkb = (__bf16*)(ws + 1 * WBYTES);
  __bf16* wvb = (__bf16*)(ws + 2 * WBYTES);
  __bf16* wob = (__bf16*)(ws + 3 * WBYTES);
  __bf16* Qp  = (__bf16*)(ws + 4 * WBYTES + 0 * ABYTES);
  __bf16* Kp  = (__bf16*)(ws + 4 * WBYTES + 1 * ABYTES);
  __bf16* Vt  = (__bf16*)(ws + 4 * WBYTES + 2 * ABYTES);
  __bf16* AO  = (__bf16*)(ws + 4 * WBYTES + 3 * ABYTES);

  const int WN = DMODEL * DMODEL;
  cvt_f32_to_bf16<<<WN / 256, 256, 0, stream>>>(Wq, wqb, WN);
  cvt_f32_to_bf16<<<WN / 256, 256, 0, stream>>>(Wk, wkb, WN);
  cvt_f32_to_bf16<<<WN / 256, 256, 0, stream>>>(Wv, wvb, WN);
  cvt_f32_to_bf16<<<WN / 256, 256, 0, stream>>>(Wo, wob, WN);

  const float qscale = 0.125f * 1.44269504f;  // 1/sqrt(DK) * log2(e)
  dim3 g(DMODEL / 128, MROWS / 128), blk(256);
  gemm_wmma<true, 0><<<g, blk, 0, stream>>>(query, wqb, bq, Qp, MROWS, DMODEL, DMODEL, qscale);
  gemm_wmma<true, 0><<<g, blk, 0, stream>>>(key,   wkb, bk, Kp, MROWS, DMODEL, DMODEL, 1.0f);
  gemm_wmma<true, 1><<<g, blk, 0, stream>>>(value, wvb, bv, Vt, MROWS, DMODEL, DMODEL, 1.0f);

  attn_wmma<<<BATCH * NHEAD * (SEQ / 128), 256, 0, stream>>>(Qp, Kp, Vt, AO);

  gemm_wmma<false, 2><<<g, blk, 0, stream>>>(AO, wob, bo, d_out, MROWS, DMODEL, DMODEL, 1.0f);
}
